// AEMTRNNSoftMoE_32315333935325
// MI455X (gfx1250) — matrix operand order repaired and checked by reading, not compiled, not run on hardware
//
#include <hip/hip_runtime.h>
#include <hip/hip_bf16.h>

typedef __attribute__((ext_vector_type(16))) _Float16 v16h;
typedef __attribute__((ext_vector_type(8)))  float    v8f;
typedef __attribute__((ext_vector_type(4)))  unsigned int v4u;

union AFrag { v16h h; v4u u[2]; };

#define FLAG_ACC    1
#define FLAG_RELU   2
#define FLAG_SIG    4
#define FLAG_BIAS   8
#define FLAG_OUTF16 16

__device__ __forceinline__ float apply_act(float v, int flags) {
  if (flags & FLAG_RELU) v = v > 0.f ? v : 0.f;
  if (flags & FLAG_SIG)  v = 1.f / (1.f + expf(-v));
  return v;
}

// Async copy of 16 bytes global -> LDS (ASYNCcnt-tracked, no VGPR round trip).
__device__ __forceinline__ void async_g2l_b128(const _Float16* g, _Float16* l) {
  unsigned lds = (unsigned)(size_t)l;                  // low 32b of generic ptr = LDS offset
  unsigned long long ga = (unsigned long long)(size_t)g;
  asm volatile("global_load_async_to_lds_b128 %0, %1, off" :: "v"(lds), "v"(ga) : "memory");
}
__device__ __forceinline__ void wait_async0() {
  asm volatile("s_wait_asynccnt 0x0" ::: "memory");
}

// ---------------------------------------------------------------------------
// Tiled WMMA GEMM:  C[z] = act( A[z] (MxK) * Bp[z] + bias[z] ),  C is f32.
// A: f16 or f32 (a_f16). Bp: f16 packed [N][K] (k-contiguous).
// Tile 64x64, 4 waves, K-step 32. Interior tiles use global_load_async_to_lds;
// edge tiles fall back to guarded scalar loads with zero fill.
// LDS pitch 40 halves keeps every fragment a 16B-aligned b128.
// ---------------------------------------------------------------------------
__global__ __launch_bounds__(128)
void gemm_wmma(const void* __restrict__ Ain, int a_f16, int lda, long aStrideZ,
               const _Float16* __restrict__ Bp, long bStrideZ,
               const float* __restrict__ bias, long biasStrideZ,
               float* __restrict__ C, int ldc, long cStrideZ,
               int M, int N, int K, int flags)
{
  __shared__ __align__(16) _Float16 As[64 * 40];
  __shared__ __align__(16) _Float16 Bs[64 * 40];
  int z = blockIdx.z;
  const _Float16* Ah = (const _Float16*)Ain + (a_f16 ? (size_t)z * aStrideZ : 0);
  const float*    Af = (const float*)Ain    + (a_f16 ? 0 : (size_t)z * aStrideZ);
  Bp += (size_t)z * bStrideZ;
  C  += (size_t)z * cStrideZ;
  if (bias) bias += (size_t)z * biasStrideZ;

  int n0 = blockIdx.x * 64, m0 = blockIdx.y * 64;
  int tid = threadIdx.x, lane = tid & 31, wv = tid >> 5;
  int hi = lane >> 4, l16 = lane & 15;

  v8f acc[4];
  for (int t = 0; t < 4; ++t) acc[t] = {};

  for (int kt = 0; kt < K; kt += 32) {
    bool a_async = a_f16 && (m0 + 64 <= M) && (kt + 32 <= K);
    bool b_async = (n0 + 64 <= N) && (kt + 32 <= K);
    if (a_async) {
      for (int i = 0; i < 2; ++i) {                    // 256 chunks of 16B
        int c = i * 128 + tid; int m = c >> 2, k0 = (c & 3) * 8;
        async_g2l_b128(Ah + (size_t)(m0 + m) * lda + kt + k0, &As[m * 40 + k0]);
      }
    } else {
      for (int i = 0; i < 16; ++i) {
        int idx = i * 128 + tid; int m = idx >> 5, k = idx & 31;
        int gm = m0 + m, gk = kt + k;
        float v = 0.f;
        if (gm < M && gk < K)
          v = a_f16 ? (float)Ah[(size_t)gm * lda + gk] : Af[(size_t)gm * lda + gk];
        As[m * 40 + k] = (_Float16)v;
      }
    }
    if (b_async) {
      for (int i = 0; i < 2; ++i) {
        int c = i * 128 + tid; int n = c >> 2, k0 = (c & 3) * 8;
        async_g2l_b128(Bp + (size_t)(n0 + n) * K + kt + k0, &Bs[n * 40 + k0]);
      }
    } else {
      for (int i = 0; i < 16; ++i) {
        int idx = i * 128 + tid; int n = idx >> 5, k = idx & 31;
        int gn = n0 + n, gk = kt + k;
        float v = (gn < N && gk < K) ? (float)Bp[(size_t)gn * K + gk] : 0.f;
        Bs[n * 40 + k] = (_Float16)v;
      }
    }
    if (a_async || b_async) wait_async0();
    __syncthreads();

    AFrag a;
    int mr = wv * 16 + l16;                 // A: M = lane%16, K split by lane-half
    a.u[0] = *(const v4u*)(&As[mr * 40 + hi * 8]);
    a.u[1] = *(const v4u*)(&As[mr * 40 + 16 + hi * 8]);
#pragma unroll
    for (int nt = 0; nt < 4; ++nt) {
      AFrag b;
      int nr = nt * 16 + l16;               // B: N = lane%16, K split by lane-half (same as A)
      b.u[0] = *(const v4u*)(&Bs[nr * 40 + hi * 8]);
      b.u[1] = *(const v4u*)(&Bs[nr * 40 + 16 + hi * 8]);
      acc[nt] = __builtin_amdgcn_wmma_f32_16x16x32_f16(
          false, a.h, false, b.h, (short)0, acc[nt], false, false);
    }
    __syncthreads();
  }

  for (int nt = 0; nt < 4; ++nt) {
    int gn = n0 + nt * 16 + l16;
#pragma unroll
    for (int r = 0; r < 8; ++r) {
      int gm = m0 + wv * 16 + hi * 8 + r;   // C/D: M = r + 8*(lane/16)
      if (gm < M && gn < N) {
        float v = acc[nt][r];
        if (flags & FLAG_BIAS) v += bias[gn];
        if (flags & FLAG_ACC)  v += C[(size_t)gm * ldc + gn];
        C[(size_t)gm * ldc + gn] = apply_act(v, flags);
      }
    }
  }
}

// ---------------------------------------------------------------------------
// Implicit-GEMM conv / conv_transpose, 4x4 kernel, stride 2, SAME.
// A (patches) gathered scalar with parity/bounds zero fill; B (weights,
// f16 packed [Cout][Cin*16]) uses async-to-LDS for interior tiles.
// out NHWC, f16 or f32 (FLAG_OUTF16).
// ---------------------------------------------------------------------------
__global__ __launch_bounds__(128)
void conv_wmma(const void* __restrict__ in, int in_f16,
               const _Float16* __restrict__ wp, const float* __restrict__ bias,
               void* __restrict__ out,
               int Bn, int Cin, int IH, int IW, int Cout, int OH, int OW,
               int in_nchw, int transposed, int flags)
{
  __shared__ __align__(16) _Float16 As[64 * 40];
  __shared__ __align__(16) _Float16 Bs[64 * 40];
  int M = Bn * OH * OW, N = Cout, K = Cin * 16;
  int n0 = blockIdx.x * 64, m0 = blockIdx.y * 64;
  int tid = threadIdx.x, lane = tid & 31, wv = tid >> 5;
  int hi = lane >> 4, l16 = lane & 15;

  v8f acc[4];
  for (int t = 0; t < 4; ++t) acc[t] = {};

  for (int kt = 0; kt < K; kt += 32) {
    for (int i = 0; i < 16; ++i) {
      int idx = i * 128 + tid; int m = idx >> 5, k = idx & 31;
      int gm = m0 + m, gk = kt + k;
      float v = 0.f;
      if (gm < M && gk < K) {
        int b = gm / (OH * OW); int rem = gm - b * OH * OW;
        int oy = rem / OW; int ox = rem - oy * OW;
        int ci = gk >> 4; int kk = gk & 15; int ky = kk >> 2; int kx = kk & 3;
        int iy, ix; bool ok;
        if (!transposed) {
          iy = oy * 2 + ky - 1; ix = ox * 2 + kx - 1;
          ok = (iy >= 0) & (iy < IH) & (ix >= 0) & (ix < IW);
        } else {
          int ty = oy + 1 - ky, tx = ox + 1 - kx;
          iy = ty >> 1; ix = tx >> 1;
          ok = (ty >= 0) & (tx >= 0) & ((ty & 1) == 0) & ((tx & 1) == 0) &
               (iy < IH) & (ix < IW);
        }
        if (ok) {
          size_t off = in_nchw ? (((size_t)b * Cin + ci) * IH + iy) * IW + ix
                               : (((size_t)b * IH + iy) * IW + ix) * Cin + ci;
          v = in_f16 ? (float)((const _Float16*)in)[off] : ((const float*)in)[off];
        }
      }
      As[m * 40 + k] = (_Float16)v;
    }
    bool b_async = (n0 + 64 <= N) && (kt + 32 <= K);
    if (b_async) {
      for (int i = 0; i < 2; ++i) {
        int c = i * 128 + tid; int n = c >> 2, k0 = (c & 3) * 8;
        async_g2l_b128(wp + (size_t)(n0 + n) * K + kt + k0, &Bs[n * 40 + k0]);
      }
    } else {
      for (int i = 0; i < 16; ++i) {
        int idx = i * 128 + tid; int n = idx >> 5, k = idx & 31;
        int gn = n0 + n, gk = kt + k;
        float v = (gn < N && gk < K) ? (float)wp[(size_t)gn * K + gk] : 0.f;
        Bs[n * 40 + k] = (_Float16)v;
      }
    }
    if (b_async) wait_async0();
    __syncthreads();

    AFrag a;
    int mr = wv * 16 + l16;
    a.u[0] = *(const v4u*)(&As[mr * 40 + hi * 8]);
    a.u[1] = *(const v4u*)(&As[mr * 40 + 16 + hi * 8]);
#pragma unroll
    for (int nt = 0; nt < 4; ++nt) {
      AFrag b;
      int nr = nt * 16 + l16;
      b.u[0] = *(const v4u*)(&Bs[nr * 40 + hi * 8]);
      b.u[1] = *(const v4u*)(&Bs[nr * 40 + 16 + hi * 8]);
      acc[nt] = __builtin_amdgcn_wmma_f32_16x16x32_f16(
          false, a.h, false, b.h, (short)0, acc[nt], false, false);
    }
    __syncthreads();
  }

  for (int nt = 0; nt < 4; ++nt) {
    int gn = n0 + nt * 16 + l16;
#pragma unroll
    for (int r = 0; r < 8; ++r) {
      int gm = m0 + wv * 16 + hi * 8 + r;
      if (gm < M && gn < N) {
        float v = acc[nt][r];
        if (flags & FLAG_BIAS) v += bias[gn];
        v = apply_act(v, flags);
        if (flags & FLAG_OUTF16) ((_Float16*)out)[(size_t)gm * N + gn] = (_Float16)v;
        else                     ((float*)out)[(size_t)gm * N + gn] = v;
      }
    }
  }
}

// ---------------------------------------------------------------------------
// Pack / convert / elementwise kernels
// ---------------------------------------------------------------------------
__global__ void k_cvt_f16(const float* __restrict__ in, _Float16* __restrict__ out, long n)
{
  long idx = (long)blockIdx.x * 256 + threadIdx.x;
  if (idx < n) out[idx] = (_Float16)in[idx];
}

// in: [z][K][N] f32  ->  out: [z][N][K] f16   (k-contiguous for async tiles)
__global__ void k_pack_bt(const float* __restrict__ in, _Float16* __restrict__ out,
                          int K, int N)
{
  int z = blockIdx.z; long nk = (long)N * K;
  long idx = (long)blockIdx.x * 256 + threadIdx.x;
  if (idx >= nk) return;
  int n = (int)(idx / K), k = (int)(idx - (long)n * K);
  out[z * nk + idx] = (_Float16)in[z * nk + (long)k * N + n];
}

__global__ void k_concat2_h(const float* __restrict__ a, int na,
                            const float* __restrict__ b, int nb,
                            _Float16* __restrict__ out, int rows)
{
  int idx = blockIdx.x * blockDim.x + threadIdx.x;
  int w = na + nb;
  if (idx >= rows * w) return;
  int r = idx / w, c = idx - r * w;
  out[idx] = (_Float16)((c < na) ? a[r * na + c] : b[r * nb + (c - na)]);
}

__global__ void k_nhwc_to_nchw(const void* __restrict__ in, void* __restrict__ out,
                               int Bn, int H, int W, int C, int f16)
{
  int idx = blockIdx.x * blockDim.x + threadIdx.x;
  int total = Bn * H * W * C;
  if (idx >= total) return;
  int c = idx % C; int t = idx / C;
  int x = t % W; t /= W;
  int y = t % H; int b = t / H;
  size_t o = (((size_t)b * C + c) * H + y) * W + x;
  if (f16) ((_Float16*)out)[o] = ((const _Float16*)in)[idx];
  else     ((float*)out)[o]    = ((const float*)in)[idx];
}

__global__ void k_rnn_combine(const float* __restrict__ pre,
                              const float* __restrict__ rnn,
                              float* __restrict__ st, int slot, float inv_tau)
{
  int idx = blockIdx.x * blockDim.x + threadIdx.x;
  if (idx >= 512 * 16 * 64) return;
  int h = idx & 63; int t = idx >> 6;          // t = b*16 + e
  size_t so = ((size_t)t * 3 + slot) * 64 + h;
  float hp = rnn[so];
  st[so] = tanhf((1.f - inv_tau) * hp + pre[idx] * inv_tau);
}

__global__ void k_moe(const float* __restrict__ states,   // [B,E,3,64]
                      const float* __restrict__ W_err, const float* __restrict__ b_err,
                      float* __restrict__ weight_out, float* __restrict__ hidden)
{
  int b = blockIdx.x, t = threadIdx.x;   // 64 threads
  __shared__ float sErr[16];
  __shared__ float sW[16];
  const float* ns = states + (size_t)b * 3072;
  if (t < 16) {
    float s = 0.f;
    for (int h = 0; h < 64; ++h) s += ns[t * 192 + 128 + h] * W_err[t * 64 + h];
    s += b_err[t];
    sErr[t] = s > 0.f ? s : 0.f;
  }
  __syncthreads();
  if (t < 16) {
    float m = -1e30f;
    for (int e = 0; e < 16; ++e) { float zz = -sErr[e]; m = zz > m ? zz : m; }
    float sum = 0.f;
    for (int e = 0; e < 16; ++e) sum += expf(-sErr[e] - m);
    float w = expf(-sErr[t] - m) / sum;
    sW[t] = w;
    weight_out[b * 16 + t] = w;
  }
  __syncthreads();
  float hv = 0.f;
  for (int e = 0; e < 16; ++e) hv += sW[e] * ns[e * 192 + 128 + t];
  hidden[b * 64 + t] = hv;
}

__global__ void k_expert_decin(const float* __restrict__ states,  // [B,E,3,64]
                               const float* __restrict__ ej,      // [E,B,8]
                               _Float16* __restrict__ dec_in, int e)
{
  int idx = blockIdx.x * blockDim.x + threadIdx.x;
  if (idx >= 512 * 72) return;
  int b = idx / 72, c = idx - b * 72;
  float v = (c < 64) ? states[((size_t)(b * 16 + e) * 3 + 2) * 64 + c]
                     : ej[(size_t)e * 512 * 8 + b * 8 + (c - 64)];
  dec_in[idx] = (_Float16)v;
}

// ---------------------------------------------------------------------------
// Host-side helpers
// ---------------------------------------------------------------------------
static void gemm(hipStream_t s, const void* A, int a_f16, int lda, long asz,
                 const _Float16* Bp, long bsz, const float* bias, long bias_sz,
                 float* C, int ldc, long csz, int M, int N, int K, int flags, int Z)
{
  dim3 g((unsigned)((N + 63) / 64), (unsigned)((M + 63) / 64), (unsigned)Z);
  gemm_wmma<<<g, dim3(128), 0, s>>>(A, a_f16, lda, asz, Bp, bsz, bias, bias_sz,
                                    C, ldc, csz, M, N, K, flags);
}

static void conv(hipStream_t s, const void* in, int in_f16, const _Float16* wp,
                 const float* bias, void* out, int Bn, int Cin, int IH, int IW,
                 int Cout, int OH, int OW, int in_nchw, int transposed, int flags)
{
  int M = Bn * OH * OW;
  dim3 g((unsigned)((Cout + 63) / 64), (unsigned)((M + 63) / 64), 1);
  conv_wmma<<<g, dim3(128), 0, s>>>(in, in_f16, wp, bias, out, Bn, Cin, IH, IW,
                                    Cout, OH, OW, in_nchw, transposed, flags);
}

static inline dim3 ew(long n) { return dim3((unsigned)((n + 255) / 256)); }

struct DecBufs {
  float* dfc; _Float16 *d1, *d2, *d3; float* d4;
  const _Float16 *fcp, *w0, *w1, *w2, *w3;
  const float *fcb, *b0, *b1, *b2, *b3;
};

static void decode_batch(hipStream_t s, const _Float16* decin, const DecBufs& d,
                         float* out_nchw)
{
  gemm(s, decin, 1, 72, 0, d.fcp, 0, d.fcb, 0, d.dfc, 2048, 0,
       512, 2048, 72, FLAG_BIAS | FLAG_RELU, 1);
  conv(s, d.dfc, 0, d.w0, d.b0, d.d1, 512, 128, 4, 4, 64, 8, 8, 1, 1,
       FLAG_BIAS | FLAG_RELU | FLAG_OUTF16);
  conv(s, d.d1, 1, d.w1, d.b1, d.d2, 512, 64, 8, 8, 32, 16, 16, 0, 1,
       FLAG_BIAS | FLAG_RELU | FLAG_OUTF16);
  conv(s, d.d2, 1, d.w2, d.b2, d.d3, 512, 32, 16, 16, 16, 32, 32, 0, 1,
       FLAG_BIAS | FLAG_RELU | FLAG_OUTF16);
  conv(s, d.d3, 1, d.w3, d.b3, d.d4, 512, 16, 32, 32, 3, 64, 64, 0, 1,
       FLAG_BIAS | FLAG_SIG);
  k_nhwc_to_nchw<<<ew(512L * 64 * 64 * 3), 256, 0, s>>>(d.d4, out_nchw, 512, 64, 64, 3, 0);
}

extern "C" void kernel_launch(void* const* d_in, const int* in_sizes, int n_in,
                              void* d_out, int out_size, void* d_ws, size_t ws_size,
                              hipStream_t stream)
{
  (void)in_sizes; (void)n_in; (void)out_size; (void)ws_size;
  const float* image      = (const float*)d_in[0];
  const float* joint      = (const float*)d_in[1];
  const float* rnn        = (const float*)d_in[2];
  const float* enc_w[4]   = {(const float*)d_in[3], (const float*)d_in[5],
                             (const float*)d_in[7], (const float*)d_in[9]};
  const float* enc_b[4]   = {(const float*)d_in[4], (const float*)d_in[6],
                             (const float*)d_in[8], (const float*)d_in[10]};
  const float* enc_fc_w   = (const float*)d_in[11];
  const float* enc_fc_b   = (const float*)d_in[12];
  const float* dec_fc_w   = (const float*)d_in[13];
  const float* dec_fc_b   = (const float*)d_in[14];
  const float* dec_w[4]   = {(const float*)d_in[15], (const float*)d_in[17],
                             (const float*)d_in[19], (const float*)d_in[21]};
  const float* dec_b[4]   = {(const float*)d_in[16], (const float*)d_in[18],
                             (const float*)d_in[20], (const float*)d_in[22]};
  const float* W_x_io     = (const float*)d_in[23];
  const float* W_io_io    = (const float*)d_in[24];
  const float* W_f_io     = (const float*)d_in[25];
  const float* W_io_f     = (const float*)d_in[26];
  const float* W_f_f      = (const float*)d_in[27];
  const float* W_s_f      = (const float*)d_in[28];
  const float* W_f_s      = (const float*)d_in[29];
  const float* W_s_s      = (const float*)d_in[30];
  const float* W_err      = (const float*)d_in[31];
  const float* b_io       = (const float*)d_in[32];
  const float* b_f        = (const float*)d_in[33];
  const float* b_s        = (const float*)d_in[34];
  const float* b_err      = (const float*)d_in[35];
  const float* joint_w    = (const float*)d_in[36];
  const float* joint_b    = (const float*)d_in[37];

  float* out = (float*)d_out;
  const long o_pred = 0;
  const long o_nj   = o_pred + 512L * 3 * 64 * 64;
  const long o_st   = o_nj   + 512L * 8;
  const long o_wgt  = o_st   + 512L * 16 * 3 * 64;
  const long o_ej   = o_wgt  + 512L * 16;
  const long o_ei   = o_ej   + 16L * 512 * 8;

  // ---- workspace carve (16B aligned) ----
  char* cur = (char*)d_ws;
  auto allocF = [&](long n) { cur = (char*)(((size_t)cur + 15) & ~(size_t)15);
                              float* p = (float*)cur; cur += n * 4; return p; };
  auto allocH = [&](long n) { cur = (char*)(((size_t)cur + 15) & ~(size_t)15);
                              _Float16* p = (_Float16*)cur; cur += n * 2; return p; };
  _Float16* e1   = allocH(512L * 32 * 32 * 16);
  _Float16* e2   = allocH(512L * 16 * 16 * 32);
  _Float16* e3   = allocH(512L * 8 * 8 * 64);
  _Float16* e4   = allocH(512L * 4 * 4 * 128);
  _Float16* e4n  = allocH(512L * 2048);
  float*    feat = allocF(512L * 16);
  _Float16* xin  = allocH(512L * 24);
  _Float16* rnnh = allocH(512L * 16 * 3 * 64);
  float*    pre  = allocF(512L * 16 * 64);
  float*    hid  = allocF(512L * 64);
  _Float16* decin= allocH(512L * 72);
  DecBufs db;
  db.dfc = allocF(512L * 2048);
  db.d1  = allocH(512L * 8 * 8 * 64);
  db.d2  = allocH(512L * 16 * 16 * 32);
  db.d3  = allocH(512L * 32 * 32 * 16);
  db.d4  = allocF(512L * 64 * 64 * 3);
  // packed f16 weights
  _Float16* encw_h[4]; _Float16* decw_h[4];
  const long encw_n[4] = {16L*48, 32L*256, 64L*512, 128L*1024};
  const long decw_n[4] = {64L*2048, 32L*1024, 16L*512, 3L*256};
  for (int i = 0; i < 4; ++i) { encw_h[i] = allocH(encw_n[i]); decw_h[i] = allocH(decw_n[i]); }
  _Float16* enc_fc_p = allocH(16L * 2048);
  _Float16* dec_fc_p = allocH(2048L * 72);
  _Float16* joint_p  = allocH(8L * 64);
  _Float16* Wx_p     = allocH(16L * 64 * 24);
  _Float16* Wr_p[7];
  for (int i = 0; i < 7; ++i) Wr_p[i] = allocH(16L * 64 * 64);

  // ---- one-time f16 packing (weights; conv layout [Cout][Cin*16] is already k-contiguous) ----
  for (int i = 0; i < 4; ++i) {
    k_cvt_f16<<<ew(encw_n[i]), 256, 0, stream>>>(enc_w[i], encw_h[i], encw_n[i]);
    k_cvt_f16<<<ew(decw_n[i]), 256, 0, stream>>>(dec_w[i], decw_h[i], decw_n[i]);
  }
  k_pack_bt<<<dim3((unsigned)((2048L*16+255)/256),1,1), 256, 0, stream>>>(enc_fc_w, enc_fc_p, 2048, 16);
  k_pack_bt<<<dim3((unsigned)((2048L*72+255)/256),1,1), 256, 0, stream>>>(dec_fc_w, dec_fc_p, 72, 2048);
  k_pack_bt<<<dim3(2,1,1), 256, 0, stream>>>(joint_w, joint_p, 64, 8);
  k_pack_bt<<<dim3(6,1,16), 256, 0, stream>>>(W_x_io, Wx_p, 24, 64);
  const float* Wr[7] = {W_io_io, W_f_io, W_io_f, W_f_f, W_s_f, W_f_s, W_s_s};
  for (int i = 0; i < 7; ++i)
    k_pack_bt<<<dim3(16,1,16), 256, 0, stream>>>(Wr[i], Wr_p[i], 64, 64);
  k_cvt_f16<<<ew(512L*16*3*64), 256, 0, stream>>>(rnn, rnnh, 512L*16*3*64);

  // ---- encoder ----
  conv(stream, image, 0, encw_h[0], enc_b[0], e1, 512, 3, 64, 64, 16, 32, 32, 1, 0,
       FLAG_BIAS | FLAG_RELU | FLAG_OUTF16);
  conv(stream, e1, 1, encw_h[1], enc_b[1], e2, 512, 16, 32, 32, 32, 16, 16, 0, 0,
       FLAG_BIAS | FLAG_RELU | FLAG_OUTF16);
  conv(stream, e2, 1, encw_h[2], enc_b[2], e3, 512, 32, 16, 16, 64, 8, 8, 0, 0,
       FLAG_BIAS | FLAG_RELU | FLAG_OUTF16);
  conv(stream, e3, 1, encw_h[3], enc_b[3], e4, 512, 64, 8, 8, 128, 4, 4, 0, 0,
       FLAG_BIAS | FLAG_RELU | FLAG_OUTF16);
  k_nhwc_to_nchw<<<ew(512L * 2048), 256, 0, stream>>>(e4, e4n, 512, 4, 4, 128, 1);
  gemm(stream, e4n, 1, 2048, 0, enc_fc_p, 0, enc_fc_b, 0, feat, 16, 0,
       512, 16, 2048, FLAG_BIAS | FLAG_RELU, 1);

  k_concat2_h<<<ew(512L * 24), 256, 0, stream>>>(feat, 16, joint, 8, xin, 512);

  // ---- MTRNN (z = expert; fully async interior tiles) ----
  gemm(stream, xin, 1, 24, 0, Wx_p, 1536, b_io, 64, pre, 1024, 64,
       512, 64, 24, FLAG_BIAS, 16);
  gemm(stream, rnnh + 0, 1, 3072, 192, Wr_p[0], 4096, nullptr, 0, pre, 1024, 64,
       512, 64, 64, FLAG_ACC, 16);
  gemm(stream, rnnh + 64, 1, 3072, 192, Wr_p[1], 4096, nullptr, 0, pre, 1024, 64,
       512, 64, 64, FLAG_ACC, 16);
  k_rnn_combine<<<ew(512L * 16 * 64), 256, 0, stream>>>(pre, rnn, out + o_st, 0, 1.f / 2.0f);
  gemm(stream, rnnh + 0, 1, 3072, 192, Wr_p[2], 4096, b_f, 64, pre, 1024, 64,
       512, 64, 64, FLAG_BIAS, 16);
  gemm(stream, rnnh + 64, 1, 3072, 192, Wr_p[3], 4096, nullptr, 0, pre, 1024, 64,
       512, 64, 64, FLAG_ACC, 16);
  gemm(stream, rnnh + 128, 1, 3072, 192, Wr_p[4], 4096, nullptr, 0, pre, 1024, 64,
       512, 64, 64, FLAG_ACC, 16);
  k_rnn_combine<<<ew(512L * 16 * 64), 256, 0, stream>>>(pre, rnn, out + o_st, 1, 1.f / 5.0f);
  gemm(stream, rnnh + 64, 1, 3072, 192, Wr_p[5], 4096, b_s, 64, pre, 1024, 64,
       512, 64, 64, FLAG_BIAS, 16);
  gemm(stream, rnnh + 128, 1, 3072, 192, Wr_p[6], 4096, nullptr, 0, pre, 1024, 64,
       512, 64, 64, FLAG_ACC, 16);
  k_rnn_combine<<<ew(512L * 16 * 64), 256, 0, stream>>>(pre, rnn, out + o_st, 2, 1.f / 70.0f);

  // ---- soft-MoE ----
  k_moe<<<dim3(512), dim3(64), 0, stream>>>(out + o_st, W_err, b_err, out + o_wgt, hid);

  gemm(stream, hid, 0, 64, 0, joint_p, 0, joint_b, 0, out + o_nj, 8, 0,
       512, 8, 64, FLAG_BIAS | FLAG_RELU, 1);
  gemm(stream, out + o_st + 0, 0, 3072, 192, joint_p, 0, joint_b, 0,
       out + o_ej, 8, 512L * 8, 512, 8, 64, FLAG_BIAS | FLAG_RELU, 16);

  // ---- decodes ----
  db.fcp = dec_fc_p; db.fcb = dec_fc_b;
  db.w0 = decw_h[0]; db.b0 = dec_b[0];
  db.w1 = decw_h[1]; db.b1 = dec_b[1];
  db.w2 = decw_h[2]; db.b2 = dec_b[2];
  db.w3 = decw_h[3]; db.b3 = dec_b[3];

  k_concat2_h<<<ew(512L * 72), 256, 0, stream>>>(hid, 64, out + o_nj, 8, decin, 512);
  decode_batch(stream, decin, db, out + o_pred);

  for (int e = 0; e < 16; ++e) {
    k_expert_decin<<<ew(512L * 72), 256, 0, stream>>>(out + o_st, out + o_ej, decin, e);
    decode_batch(stream, decin, db, out + o_ei + (long)e * 512 * 3 * 64 * 64);
  }
}
